// PNA_60997125538472
// MI455X (gfx1250) — matrix operand order repaired and checked by reading, not compiled, not run on hardware
//
#include <hip/hip_runtime.h>
#include <hip/hip_bf16.h>
#include <math.h>

// ---------------- problem constants ----------------
#define Nn 25000
#define Ee 400000
#define NFd 30
#define EFd 10
#define Hh 40
#define Tt 5
#define Ff 40
#define FOd 8
#define TFd 200      // T*F, width of hs
#define K13 520      // 13*F, post-GEMM K
#define LDKB 128     // padded K for edge pre GEMM (120 -> 128)
#define NCB 208      // padded cols for edge pre GEMM (200 -> 208)
#define LDKC 544     // padded K for node post GEMM (520 -> 544)
#define LDK1 128     // padded K for mlp1 (120 -> 128)

// f16 LDS-image sizes (halves)
#define IMG_PRE   (NCB * LDKB)     // 26624 halves = 53248 B
#define IMG_POST  (80 * LDKC)      // 43520 halves = 87040 B
#define IMG_LIN   (48 * 64)        // 3072 halves  = 6144 B
#define IMG_E2    (48 * 64)        // 3072 halves
#define IMG_MLP1  (64 * LDK1)      // 8192 halves
#define IMG_MLP2  (32 * 64)        // 2048 halves
#define IMG_MLP3  (16 * 32)        // 512 halves
#define IMG_MLP   (IMG_MLP1 + IMG_MLP2 + IMG_MLP3)  // 10752 halves

typedef __attribute__((ext_vector_type(16))) _Float16 v16h;
typedef __attribute__((ext_vector_type(8)))  _Float16 v8h;
typedef __attribute__((ext_vector_type(8)))  float    v8f;
typedef __attribute__((ext_vector_type(4)))  unsigned tdm_v4u;
typedef __attribute__((ext_vector_type(8)))  int      tdm_v8i;
typedef __attribute__((ext_vector_type(4)))  int      tdm_v4i;

#if defined(__has_builtin)
#if __has_builtin(__builtin_amdgcn_tensor_load_to_lds) && \
    __has_builtin(__builtin_amdgcn_s_wait_tensorcnt)
#define USE_TDM 1
#endif
#endif
#ifndef USE_TDM
#define USE_TDM 0
#endif

// ---------------- Tensor Data Mover: 1-D LDS stage of a premade f16 image ------
// D# group0: count=1, lds_addr, 57-bit global addr, type=2 ("image").
// D# group1: data_size=2 (4B units), tile_dim0 = nwords (<=16 bits), tile_dim1=1,
//            tensor_dim0 = nwords, tensor_dim0_stride = nwords.
__device__ __forceinline__ void stage_lds(_Float16* dstLds, const _Float16* srcGlb,
                                          int halves, int tid, int nthreads) {
#if USE_TDM
  if (tid < 32) {  // one wave issues the DMA; EXEC is ignored by TENSOR ops
    unsigned nwords = (unsigned)(halves >> 1);
    unsigned long long ga = (unsigned long long)(size_t)srcGlb;
    unsigned lds = (unsigned)(size_t)dstLds;  // ISA: LDS aperture uses addr[31:0]
    tdm_v4u g0 = {0u, 0u, 0u, 0u};
    g0[0] = 1u;                                   // count=1, user descriptor
    g0[1] = lds;                                  // lds_addr
    g0[2] = (unsigned)ga;                         // global_addr[31:0]
    g0[3] = (unsigned)(ga >> 32) | (2u << 30);    // global_addr[56:32] | type=2
    tdm_v8i g1 = {0, 0, 0, 0, 0, 0, 0, 0};
    g1[0] = 0x20000;                              // data_size=2 (4-byte units)
    g1[2] = (int)nwords;                          // tensor_dim0
    g1[3] = 1;                                    // tensor_dim1
    g1[4] = (int)(nwords | (1u << 16));           // tile_dim0 | tile_dim1=1
    g1[5] = (int)nwords;                          // tensor_dim0_stride[31:0]
    tdm_v4i z4 = {0, 0, 0, 0};
#if __clang_major__ >= 23
    tdm_v8i z8 = {0, 0, 0, 0, 0, 0, 0, 0};
    __builtin_amdgcn_tensor_load_to_lds(g0, g1, z4, z4, z8, 0);
#else
    __builtin_amdgcn_tensor_load_to_lds(g0, g1, z4, z4, 0);
#endif
    __builtin_amdgcn_s_wait_tensorcnt(0);
  }
#else
  v8h* d = (v8h*)dstLds;
  const v8h* s = (const v8h*)srcGlb;
  for (int i = tid; i < (halves >> 3); i += nthreads) d[i] = s[i];
#endif
}

// ---------------- WMMA helpers (wave32, 16x16x32 f16 -> f32) ----------------
__device__ __forceinline__ v8f wmma_step(v16h a, v16h b, v8f c) {
  return __builtin_amdgcn_wmma_f32_16x16x32_f16(false, a, false, b, (short)0, c,
                                                false, false);
}

// A fragment: 16x32 f16 tile from LDS, rows row-major, leading dim ldk (halves).
__device__ __forceinline__ v16h frag_a(const _Float16* A, int ldk, int kbase, int lane) {
  int row = lane & 15, hi = (lane >> 4) & 1;
  const _Float16* p = A + row * ldk + kbase;
  v8h lo = *(const v8h*)(p + hi * 8);
  v8h hv = *(const v8h*)(p + 16 + hi * 8);
  v16h a;
#pragma unroll
  for (int j = 0; j < 8; ++j) { a[j] = lo[j]; a[j + 8] = hv[j]; }
  return a;
}

// B fragment: 32x16 f16 tile; B stored TRANSPOSED in LDS: Bt[col][k].
__device__ __forceinline__ v16h frag_b(const _Float16* Bt, int ldk, int colbase,
                                       int kbase, int lane) {
  int col = lane & 15, hi = (lane >> 4) & 1;
  const _Float16* p = Bt + (colbase + col) * ldk + kbase + hi * 8;
  v8h lo = *(const v8h*)p;
  v8h hv = *(const v8h*)(p + 16);
  v16h b;
#pragma unroll
  for (int j = 0; j < 8; ++j) { b[j] = lo[j]; b[j + 8] = hv[j]; }
  return b;
}

// monotone float <-> uint mapping for atomic min/max
__device__ __forceinline__ unsigned ford(float f) {
  unsigned u = __float_as_uint(f);
  return (u & 0x80000000u) ? ~u : (u | 0x80000000u);
}
__device__ __forceinline__ float ford_inv(unsigned u) {
  unsigned v = (u & 0x80000000u) ? (u ^ 0x80000000u) : ~u;
  return __uint_as_float(v);
}

// ---------------- weight-image prep kernels (one-time, tiny) ----------------
__global__ void k_prep_pre(const float* pre_w, _Float16* img) {
  int idx = blockIdx.x * blockDim.x + threadIdx.x;
  if (idx >= IMG_PRE) return;
  int c = idx / LDKB, k = idx % LDKB;
  float v = 0.f;
  if (c < TFd && k < 120) {
    int t = c / Ff, f = c % Ff;
    v = pre_w[(t * 120 + k) * Ff + f];
  }
  img[idx] = (_Float16)v;
}

__global__ void k_prep_post(const float* post_w, const float* lin_w, _Float16* imgP,
                            _Float16* imgL) {
  int idx = blockIdx.x * blockDim.x + threadIdx.x;
  if (idx < IMG_POST) {
    int c = idx / LDKC, k = idx % LDKC;
    int t = c >> 4, f = c & 15;
    imgP[idx] =
        (_Float16)((f < FOd && k < K13) ? post_w[(t * K13 + k) * FOd + f] : 0.f);
  }
  if (idx < IMG_LIN) {
    int c = idx / 64, k = idx % 64;
    imgL[idx] = (_Float16)((c < Hh && k < Hh) ? lin_w[k * Hh + c] : 0.f);
  }
}

__global__ void k_prep_e2(const float* encw, _Float16* img) {
  int idx = blockIdx.x * blockDim.x + threadIdx.x;
  if (idx >= IMG_E2) return;
  int c = idx / 64, k = idx % 64;
  img[idx] = (_Float16)((c < Hh && k < Hh) ? encw[k * Hh + c] : 0.f);
}

__global__ void k_prep_mlp(const float* w1, const float* w2, const float* w3,
                           _Float16* img) {
  int idx = blockIdx.x * blockDim.x + threadIdx.x;
  if (idx >= IMG_MLP) return;
  if (idx < IMG_MLP1) {
    int c = idx / LDK1, k = idx % LDK1;
    img[idx] = (_Float16)((c < 50 && k < 120) ? w1[k * 50 + c] : 0.f);
  } else if (idx < IMG_MLP1 + IMG_MLP2) {
    int j = idx - IMG_MLP1;
    int c = j / 64, k = j % 64;
    img[idx] = (_Float16)((c < 25 && k < 50) ? w2[k * 25 + c] : 0.f);
  } else {
    int j = idx - IMG_MLP1 - IMG_MLP2;
    int c = j / 32, k = j % 32;
    img[idx] = (_Float16)((c < 2 && k < 25) ? w3[k * 2 + c] : 0.f);
  }
}

// ---------------- small utility kernels ----------------
__global__ void k_fzero(float* p, int n) {
  int i = blockIdx.x * blockDim.x + threadIdx.x;
  if (i < n) p[i] = 0.f;
}

__global__ void k_node_enc(const float* x, const float* w, const float* b, float* h) {
  __shared__ float sw[NFd * Hh + Hh];
  for (int i = threadIdx.x; i < NFd * Hh; i += blockDim.x) sw[i] = w[i];
  for (int i = threadIdx.x; i < Hh; i += blockDim.x) sw[NFd * Hh + i] = b[i];
  __syncthreads();
  int idx = blockIdx.x * blockDim.x + threadIdx.x;
  if (idx >= Nn * Hh) return;
  int n = idx / Hh, c = idx % Hh;
  float acc = sw[NFd * Hh + c];
#pragma unroll
  for (int k = 0; k < NFd; ++k) acc += x[n * NFd + k] * sw[k * Hh + c];
  h[idx] = acc;
}

__global__ void k_edge_enc(const float* ea, const float* w, const float* b, float* e) {
  __shared__ float sw[EFd * Hh + Hh];
  for (int i = threadIdx.x; i < EFd * Hh; i += blockDim.x) sw[i] = w[i];
  for (int i = threadIdx.x; i < Hh; i += blockDim.x) sw[EFd * Hh + i] = b[i];
  __syncthreads();
  long long idx = (long long)blockIdx.x * blockDim.x + threadIdx.x;
  if (idx >= (long long)Ee * Hh) return;
  int n = (int)(idx / Hh), c = (int)(idx % Hh);
  float acc = sw[EFd * Hh + c];
#pragma unroll
  for (int k = 0; k < EFd; ++k) acc += ea[n * EFd + k] * sw[k * Hh + c];
  e[idx] = acc;
}

__global__ void k_count(const int* dst, float* cnt) {
  int e = blockIdx.x * blockDim.x + threadIdx.x;
  if (e < Ee) atomicAdd(&cnt[dst[e]], 1.0f);
}

__global__ void k_logred(const float* cnt, float* redbuf) {
  __shared__ float sred[256];
  float s = 0.f;
  for (int i = blockIdx.x * blockDim.x + threadIdx.x; i < Nn;
       i += gridDim.x * blockDim.x)
    s += logf(cnt[i] + 1.f);
  sred[threadIdx.x] = s;
  __syncthreads();
  for (int off = 128; off > 0; off >>= 1) {
    if (threadIdx.x < off) sred[threadIdx.x] += sred[threadIdx.x + off];
    __syncthreads();
  }
  if (threadIdx.x == 0) atomicAdd(redbuf, sred[0]);
}

__global__ void k_ampatt(const float* cnt, const float* redbuf, float* amp, float* att) {
  int n = blockIdx.x * blockDim.x + threadIdx.x;
  if (n >= Nn) return;
  float avg = redbuf[0] / (float)Nn;
  float c1 = fmaxf(cnt[n], 1.f);
  float ld = logf(c1 + 1.f);
  amp[n] = ld / avg;
  att[n] = avg / ld;
}

__global__ void k_initagg(float* s, float* q, unsigned* mn, unsigned* mx) {
  long long i = (long long)blockIdx.x * blockDim.x + threadIdx.x;
  if (i >= (long long)Nn * TFd) return;
  s[i] = 0.f;
  q[i] = 0.f;
  mn[i] = ford(INFINITY);
  mx[i] = ford(-INFINITY);
}

// ---------------- e2 = e_emb @ enc_w + enc_b  (WMMA) ----------------
__global__ __launch_bounds__(256) void k_e2(const float* e_emb, const _Float16* imgB,
                                            const float* encb, float* e2) {
  __shared__ __attribute__((aligned(16))) _Float16 sB[IMG_E2];
  __shared__ __attribute__((aligned(16))) _Float16 sA[8 * 16 * 64];
  stage_lds(sB, imgB, IMG_E2, threadIdx.x, 256);
  int wave = threadIdx.x >> 5, lane = threadIdx.x & 31;
  int ebase = (blockIdx.x * 8 + wave) * 16;  // exact tiling: ebase+15 < Ee always
  _Float16* A = sA + wave * 16 * 64;
  for (int idx = lane; idx < 16 * 64; idx += 32) {
    int r = idx / 64, k = idx % 64;
    A[idx] = (_Float16)((k < Hh) ? e_emb[(long long)(ebase + r) * Hh + k] : 0.f);
  }
  __syncthreads();
  v16h a0 = frag_a(A, 64, 0, lane);
  v16h a1 = frag_a(A, 64, 32, lane);
  int col = lane & 15, hi = lane >> 4;
  for (int ct = 0; ct < 3; ++ct) {
    v8f acc = {};
    acc = wmma_step(a0, frag_b(sB, 64, ct * 16, 0, lane), acc);
    acc = wmma_step(a1, frag_b(sB, 64, ct * 16, 32, lane), acc);
    int gc = ct * 16 + col;
    if (gc < Hh) {
      float bias = encb[gc];
#pragma unroll
      for (int r = 0; r < 8; ++r)
        e2[(long long)(ebase + r + hi * 8) * Hh + gc] = acc[r] + bias;
    }
  }
}

// ---------------- fused edge tower GEMM + segment scatter ----------------
__global__ __launch_bounds__(256) void k_edge_pre(
    const float* __restrict__ h, const float* __restrict__ e2,
    const int* __restrict__ src, const int* __restrict__ dst,
    const _Float16* __restrict__ imgB, const float* __restrict__ pre_b,
    float* aggsum, float* aggsq, unsigned* aggmin, unsigned* aggmax) {
  __shared__ __attribute__((aligned(16))) _Float16 sB[IMG_PRE];        // 53 KB
  __shared__ __attribute__((aligned(16))) _Float16 sA[8 * 16 * LDKB];  // 32 KB
  __shared__ int sDst[8 * 16];
  stage_lds(sB, imgB, IMG_PRE, threadIdx.x, 256);
  int wave = threadIdx.x >> 5, lane = threadIdx.x & 31;
  int ebase = (blockIdx.x * 8 + wave) * 16;  // exact tiling over Ee
  _Float16* A = sA + wave * 16 * LDKB;
  for (int idx = lane; idx < 16 * LDKB; idx += 32) {
    int r = idx / LDKB, k = idx % LDKB;
    int e = ebase + r;
    float v = 0.f;
    if (k < 120) {
      if (k < 40)      v = h[dst[e] * Hh + k];
      else if (k < 80) v = h[src[e] * Hh + (k - 40)];
      else             v = e2[(long long)e * Hh + (k - 80)];
      if (k == 0 && e + 128 < Ee)  // stream-ahead hint -> global_prefetch_b8
        __builtin_prefetch(&e2[(long long)(e + 128) * Hh], 0, 1);
    }
    A[idx] = (_Float16)v;
  }
  if (lane < 16) sDst[wave * 16 + lane] = dst[ebase + lane];
  __syncthreads();
  v16h a0 = frag_a(A, LDKB, 0, lane);
  v16h a1 = frag_a(A, LDKB, 32, lane);
  v16h a2 = frag_a(A, LDKB, 64, lane);
  v16h a3 = frag_a(A, LDKB, 96, lane);
  int col = lane & 15, hi = lane >> 4;
  for (int ct = 0; ct < 13; ++ct) {
    v8f acc = {};
    acc = wmma_step(a0, frag_b(sB, LDKB, ct * 16, 0, lane), acc);
    acc = wmma_step(a1, frag_b(sB, LDKB, ct * 16, 32, lane), acc);
    acc = wmma_step(a2, frag_b(sB, LDKB, ct * 16, 64, lane), acc);
    acc = wmma_step(a3, frag_b(sB, LDKB, ct * 16, 96, lane), acc);
    int gc = ct * 16 + col;
    if (gc < TFd) {
      int t = gc / Ff, f = gc % Ff;
      float bias = pre_b[t * Ff + f];
#pragma unroll
      for (int r = 0; r < 8; ++r) {
        int row = r + hi * 8;
        int d = sDst[wave * 16 + row];
        float v = acc[r] + bias;
        long long off = (long long)d * TFd + gc;
        atomicAdd(&aggsum[off], v);
        atomicAdd(&aggsq[off], v * v);
        unsigned u = ford(v);
        atomicMin(&aggmin[off], u);
        atomicMax(&aggmax[off], u);
      }
    }
  }
}

// finalize in place: sum->mean, sq->std, umin/umax -> float mn/mx
__global__ void k_aggfin(const float* cnt, float* sum, float* sq, unsigned* umin,
                         unsigned* umax) {
  long long idx = (long long)blockIdx.x * blockDim.x + threadIdx.x;
  if (idx >= (long long)Nn * TFd) return;
  int n = (int)(idx / TFd);
  float c = cnt[n], c1 = fmaxf(c, 1.f);
  float mean = sum[idx] / c1;
  float msq = sq[idx] / c1;
  float sd = sqrtf(fmaxf(msq - mean * mean, 0.f) + 1e-5f);
  float mn = (c > 0.f) ? ford_inv(umin[idx]) : 0.f;
  float mx = (c > 0.f) ? ford_inv(umax[idx]) : 0.f;
  sum[idx] = mean;
  sq[idx] = sd;
  ((float*)umin)[idx] = mn;
  ((float*)umax)[idx] = mx;
}

// ---------------- node-side post GEMM (K=520 per tower) + lin + BN stats ---------
__global__ __launch_bounds__(64) void k_node_post(
    const float* __restrict__ h, const float* __restrict__ meanb,
    const float* __restrict__ mnb, const float* __restrict__ mxb,
    const float* __restrict__ stdb, const float* __restrict__ amp,
    const float* __restrict__ att, const _Float16* __restrict__ imgPost,
    const _Float16* __restrict__ imgLin, const float* __restrict__ post_b,
    const float* __restrict__ lin_b, float* cbuf, float* bnsum, float* bnsq) {
  __shared__ __attribute__((aligned(16))) _Float16 sBpost[IMG_POST];    // 87 KB
  __shared__ __attribute__((aligned(16))) _Float16 sA[2 * 16 * LDKC];   // 34 KB
  __shared__ float sPre[2 * 16 * 48];                                   // 6 KB
  __shared__ __attribute__((aligned(16))) _Float16 sAlin[2 * 16 * 64];  // 4 KB
  __shared__ __attribute__((aligned(16))) _Float16 sBlin[IMG_LIN];      // 6 KB
  stage_lds(sBpost, imgPost, IMG_POST, threadIdx.x, 64);
  stage_lds(sBlin, imgLin, IMG_LIN, threadIdx.x, 64);
  int wave = threadIdx.x >> 5, lane = threadIdx.x & 31;
  int nbase = (blockIdx.x * 2 + wave) * 16;
  _Float16* A = sA + wave * 16 * LDKC;
  float* Pre = sPre + wave * 16 * 48;
  int col = lane & 15, hi = lane >> 4;
  const float* stats[4] = {meanb, mnb, mxb, stdb};
  __syncthreads();

  for (int t = 0; t < Tt; ++t) {
    // 520-wide per-tower A rows: [x(40) | agg(160) | agg*amp | agg*att]
    for (int idx = lane; idx < 16 * LDKC; idx += 32) {
      int r = idx / LDKC, k = idx % LDKC;
      int n = nbase + r;
      float v = 0.f;
      if (n < Nn && k < K13) {
        if (k < Hh) {
          v = h[n * Hh + k];
        } else {
          int k2 = k - Hh;
          int s = k2 / 160, k3 = k2 % 160;
          int st = k3 / Ff, f = k3 % Ff;
          v = stats[st][(long long)n * TFd + t * Ff + f];
          if (s == 1) v *= amp[n];
          else if (s == 2) v *= att[n];
        }
      }
      A[idx] = (_Float16)v;
    }
    __syncthreads();
    v8f acc = {};
#pragma unroll
    for (int ks = 0; ks < 17; ++ks)
      acc = wmma_step(frag_a(A, LDKC, ks * 32, lane),
                      frag_b(sBpost, LDKC, t * 16, ks * 32, lane), acc);
    if (col < FOd) {
      float bias = post_b[t * FOd + col];
#pragma unroll
      for (int r = 0; r < 8; ++r)
        Pre[(r + hi * 8) * 48 + t * FOd + col] = acc[r] + bias;
    }
    __syncthreads();
  }

  // lin: (16x40) @ (40x40)
  _Float16* Alin = sAlin + wave * 16 * 64;
  for (int idx = lane; idx < 16 * 64; idx += 32) {
    int r = idx / 64, k = idx % 64;
    Alin[idx] = (_Float16)((k < Hh) ? Pre[r * 48 + k] : 0.f);
  }
  __syncthreads();
  v16h la0 = frag_a(Alin, 64, 0, lane);
  v16h la1 = frag_a(Alin, 64, 32, lane);
  for (int ct = 0; ct < 3; ++ct) {
    v8f acc = {};
    acc = wmma_step(la0, frag_b(sBlin, 64, ct * 16, 0, lane), acc);
    acc = wmma_step(la1, frag_b(sBlin, 64, ct * 16, 32, lane), acc);
    int gc = ct * 16 + col;
    if (gc < Hh) {
      float bias = lin_b[gc];
      float lsum = 0.f, lsq = 0.f;
#pragma unroll
      for (int r = 0; r < 8; ++r) {
        int n = nbase + r + hi * 8;
        if (n < Nn) {
          float v = acc[r] + bias;
          cbuf[n * Hh + gc] = v;
          lsum += v;
          lsq += v * v;
        }
      }
      atomicAdd(&bnsum[gc], lsum);
      atomicAdd(&bnsq[gc], lsq);
    }
  }
}

__global__ void k_bnapply(const float* cbuf, const float* bnsum, const float* bnsq,
                          const float* g, const float* b, float* h) {
  int idx = blockIdx.x * blockDim.x + threadIdx.x;
  if (idx >= Nn * Hh) return;
  int c = idx % Hh;
  float mu = bnsum[c] / (float)Nn;
  float var = bnsq[c] / (float)Nn - mu * mu;
  float v = (cbuf[idx] - mu) * rsqrtf(var + 1e-5f) * g[c] + b[c];
  h[idx] = (h[idx] + fmaxf(v, 0.f)) * 0.5f;
}

// ---------------- final edge MLP: 120 -> 50 -> 25 -> 2, all WMMA -------------
__global__ __launch_bounds__(256) void k_mlp(
    const float* __restrict__ h, const float* __restrict__ e_emb,
    const int* __restrict__ src, const int* __restrict__ dst,
    const _Float16* __restrict__ imgB, const float* __restrict__ b1,
    const float* __restrict__ b2, const float* __restrict__ b3, float* out) {
  __shared__ __attribute__((aligned(16))) _Float16 sBall[IMG_MLP];     // 21 KB
  __shared__ __attribute__((aligned(16))) _Float16 sA[8 * 16 * LDK1];  // 32 KB
  __shared__ __attribute__((aligned(16))) _Float16 sZ1[8 * 16 * 64];   // 16 KB
  __shared__ __attribute__((aligned(16))) _Float16 sZ2[8 * 16 * 32];   // 8 KB
  stage_lds(sBall, imgB, IMG_MLP, threadIdx.x, 256);
  const _Float16* sB1 = sBall;
  const _Float16* sB2 = sBall + IMG_MLP1;
  const _Float16* sB3 = sBall + IMG_MLP1 + IMG_MLP2;
  int wave = threadIdx.x >> 5, lane = threadIdx.x & 31;
  int ebase = (blockIdx.x * 8 + wave) * 16;  // exact tiling over Ee
  _Float16* A = sA + wave * 16 * LDK1;
  _Float16* Z1 = sZ1 + wave * 16 * 64;
  _Float16* Z2 = sZ2 + wave * 16 * 32;
  for (int idx = lane; idx < 16 * LDK1; idx += 32) {
    int r = idx / LDK1, k = idx % LDK1;
    int e = ebase + r;
    float v = 0.f;
    if (k < 120) {
      if (k < 40)      v = fmaxf(h[src[e] * Hh + k], 0.f);
      else if (k < 80) v = fmaxf(h[dst[e] * Hh + (k - 40)], 0.f);
      else             v = e_emb[(long long)e * Hh + (k - 80)];
    }
    A[idx] = (_Float16)v;
  }
  __syncthreads();
  int col = lane & 15, hi = lane >> 4;
  v16h a0 = frag_a(A, LDK1, 0, lane);
  v16h a1 = frag_a(A, LDK1, 32, lane);
  v16h a2 = frag_a(A, LDK1, 64, lane);
  v16h a3 = frag_a(A, LDK1, 96, lane);
  for (int ct = 0; ct < 4; ++ct) {
    v8f acc = {};
    acc = wmma_step(a0, frag_b(sB1, LDK1, ct * 16, 0, lane), acc);
    acc = wmma_step(a1, frag_b(sB1, LDK1, ct * 16, 32, lane), acc);
    acc = wmma_step(a2, frag_b(sB1, LDK1, ct * 16, 64, lane), acc);
    acc = wmma_step(a3, frag_b(sB1, LDK1, ct * 16, 96, lane), acc);
    int gc = ct * 16 + col;
    float bias = (gc < 50) ? b1[gc] : 0.f;
#pragma unroll
    for (int r = 0; r < 8; ++r)
      Z1[(r + hi * 8) * 64 + gc] = (_Float16)fmaxf(acc[r] + bias, 0.f);
  }
  __syncthreads();
  v16h z0 = frag_a(Z1, 64, 0, lane);
  v16h z1 = frag_a(Z1, 64, 32, lane);
  for (int ct = 0; ct < 2; ++ct) {
    v8f acc = {};
    acc = wmma_step(z0, frag_b(sB2, 64, ct * 16, 0, lane), acc);
    acc = wmma_step(z1, frag_b(sB2, 64, ct * 16, 32, lane), acc);
    int gc = ct * 16 + col;
    float bias = (gc < 25) ? b2[gc] : 0.f;
#pragma unroll
    for (int r = 0; r < 8; ++r)
      Z2[(r + hi * 8) * 32 + gc] = (_Float16)fmaxf(acc[r] + bias, 0.f);
  }
  __syncthreads();
  {
    v8f acc = {};
    acc = wmma_step(frag_a(Z2, 32, 0, lane), frag_b(sB3, 32, 0, 0, lane), acc);
    if (col < 2) {
      float bias = b3[col];
#pragma unroll
      for (int r = 0; r < 8; ++r)
        out[(long long)(ebase + r + hi * 8) * 2 + col] = acc[r] + bias;
    }
  }
}

// ---------------- launcher ----------------
extern "C" void kernel_launch(void* const* d_in, const int* in_sizes, int n_in,
                              void* d_out, int out_size, void* d_ws, size_t ws_size,
                              hipStream_t stream) {
  const float* x         = (const float*)d_in[0];
  const float* edge_attr = (const float*)d_in[1];
  const int*   eidx      = (const int*)d_in[2];
  const int*   src       = eidx;
  const int*   dst       = eidx + Ee;
  const float* node_w    = (const float*)d_in[3];
  const float* node_b    = (const float*)d_in[4];
  const float* edge_w    = (const float*)d_in[5];
  const float* edge_b    = (const float*)d_in[6];
  const float* enc_w     = (const float*)d_in[7];
  const float* enc_b     = (const float*)d_in[8];
  const float* pre_w     = (const float*)d_in[9];
  const float* pre_b     = (const float*)d_in[10];
  const float* post_w    = (const float*)d_in[11];
  const float* post_b    = (const float*)d_in[12];
  const float* lin_w     = (const float*)d_in[13];
  const float* lin_b     = (const float*)d_in[14];
  const float* bn_g      = (const float*)d_in[15];
  const float* bn_b      = (const float*)d_in[16];
  const float* mlp1_w    = (const float*)d_in[17];
  const float* mlp1_b    = (const float*)d_in[18];
  const float* mlp2_w    = (const float*)d_in[19];
  const float* mlp2_b    = (const float*)d_in[20];
  const float* mlp3_w    = (const float*)d_in[21];
  const float* mlp3_b    = (const float*)d_in[22];
  float* out = (float*)d_out;

  char* p = (char*)d_ws;
  auto take = [&](size_t bytes) {
    void* r = (void*)p;
    p += (bytes + 255) & ~(size_t)255;
    return r;
  };
  float*     h       = (float*)take(sizeof(float) * Nn * Hh);
  float*     e_emb   = (float*)take(sizeof(float) * (size_t)Ee * Hh);
  float*     e2      = (float*)take(sizeof(float) * (size_t)Ee * Hh);
  float*     cnt     = (float*)take(sizeof(float) * Nn);
  float*     amp     = (float*)take(sizeof(float) * Nn);
  float*     att     = (float*)take(sizeof(float) * Nn);
  float*     redbuf  = (float*)take(sizeof(float) * 1);
  float*     aggsum  = (float*)take(sizeof(float) * (size_t)Nn * TFd);
  float*     aggsq   = (float*)take(sizeof(float) * (size_t)Nn * TFd);
  unsigned*  aggmin  = (unsigned*)take(sizeof(unsigned) * (size_t)Nn * TFd);
  unsigned*  aggmax  = (unsigned*)take(sizeof(unsigned) * (size_t)Nn * TFd);
  float*     cbuf    = (float*)take(sizeof(float) * Nn * Hh);
  float*     bnacc   = (float*)take(sizeof(float) * 2 * Hh);  // [sum | sq]
  _Float16*  imgPre  = (_Float16*)take(sizeof(_Float16) * IMG_PRE);
  _Float16*  imgPost = (_Float16*)take(sizeof(_Float16) * IMG_POST);
  _Float16*  imgLin  = (_Float16*)take(sizeof(_Float16) * IMG_LIN);
  _Float16*  imgE2   = (_Float16*)take(sizeof(_Float16) * IMG_E2);
  _Float16*  imgMlp  = (_Float16*)take(sizeof(_Float16) * IMG_MLP);

  const int TB = 256;
  auto cdiv = [](long long a, long long b) { return (int)((a + b - 1) / b); };

  // encoders
  k_node_enc<<<cdiv((long long)Nn * Hh, TB), TB, 0, stream>>>(x, node_w, node_b, h);
  k_edge_enc<<<cdiv((long long)Ee * Hh, TB), TB, 0, stream>>>(edge_attr, edge_w,
                                                              edge_b, e_emb);
  // degree statistics
  k_fzero<<<cdiv(Nn, TB), TB, 0, stream>>>(cnt, Nn);
  k_fzero<<<1, TB, 0, stream>>>(redbuf, 1);
  k_count<<<cdiv(Ee, TB), TB, 0, stream>>>(dst, cnt);
  k_logred<<<256, TB, 0, stream>>>(cnt, redbuf);
  k_ampatt<<<cdiv(Nn, TB), TB, 0, stream>>>(cnt, redbuf, amp, att);

  // one-time weight image for the final MLP
  k_prep_mlp<<<cdiv(IMG_MLP, TB), TB, 0, stream>>>(mlp1_w, mlp2_w, mlp3_w, imgMlp);

  const int edgeBlocks = Ee / 16 / 8;                 // 3125 (exact)
  const int nodeBlocks = cdiv(cdiv(Nn, 16), 2);       // 782

  for (int i = 0; i < 2; ++i) {
    k_prep_e2<<<cdiv(IMG_E2, TB), TB, 0, stream>>>(enc_w + i * Hh * Hh, imgE2);
    k_prep_pre<<<cdiv(IMG_PRE, TB), TB, 0, stream>>>(pre_w + i * Tt * 120 * Ff,
                                                     imgPre);
    k_prep_post<<<cdiv(IMG_POST, TB), TB, 0, stream>>>(
        post_w + i * Tt * K13 * FOd, lin_w + i * Hh * Hh, imgPost, imgLin);
    k_e2<<<edgeBlocks, TB, 0, stream>>>(e_emb, imgE2, enc_b + i * Hh, e2);
    k_initagg<<<cdiv((long long)Nn * TFd, TB), TB, 0, stream>>>(aggsum, aggsq,
                                                                aggmin, aggmax);
    k_fzero<<<1, TB, 0, stream>>>(bnacc, 2 * Hh);
    k_edge_pre<<<edgeBlocks, TB, 0, stream>>>(h, e2, src, dst, imgPre,
                                              pre_b + i * Tt * Ff, aggsum, aggsq,
                                              aggmin, aggmax);
    k_aggfin<<<cdiv((long long)Nn * TFd, TB), TB, 0, stream>>>(cnt, aggsum, aggsq,
                                                               aggmin, aggmax);
    k_node_post<<<nodeBlocks, 64, 0, stream>>>(
        h, aggsum, (const float*)aggmin, (const float*)aggmax, aggsq, amp, att,
        imgPost, imgLin, post_b + i * Tt * FOd, lin_b + i * Hh, cbuf, bnacc,
        bnacc + Hh);
    k_bnapply<<<cdiv((long long)Nn * Hh, TB), TB, 0, stream>>>(
        cbuf, bnacc, bnacc + Hh, bn_g + i * Hh, bn_b + i * Hh, h);
  }

  k_mlp<<<edgeBlocks, TB, 0, stream>>>(h, e_emb, src, dst, imgMlp, mlp1_b, mlp2_b,
                                       mlp3_b, out);
}
